// TransformerAttention_45071386804873
// MI455X (gfx1250) — compile-verified, matrix operand order checked by reading, not run on hardware
//
#include <hip/hip_runtime.h>
#include <hip/hip_bf16.h>

// ---------------------------------------------------------------------------
// Types
// ---------------------------------------------------------------------------
typedef __bf16 bf16_t;
typedef __attribute__((ext_vector_type(16))) __bf16 v16bf;
typedef __attribute__((ext_vector_type(8)))  __bf16 v8bf;
typedef __attribute__((ext_vector_type(8)))  float  v8f;
typedef __attribute__((vector_size(16)))     int    v4i_t;   // matches builtin proto

#ifndef __has_builtin
#define __has_builtin(x) 0
#endif
#if __has_builtin(__builtin_amdgcn_global_load_async_to_lds_b128) && \
    __has_builtin(__builtin_amdgcn_s_wait_asynccnt)
#define HAVE_ASYNC_LDS 1
#else
#define HAVE_ASYNC_LDS 0
#endif

static __device__ inline bf16_t f2bf(float f) {
    unsigned u = __float_as_uint(f);
    u = u + 0x7FFFu + ((u >> 16) & 1u);          // round-to-nearest-even
    unsigned short h = (unsigned short)(u >> 16);
    return __builtin_bit_cast(__bf16, h);
}
static __device__ inline float bf2f(bf16_t b) {
    unsigned short h = __builtin_bit_cast(unsigned short, b);
    return __uint_as_float(((unsigned)h) << 16);
}

static __device__ inline v8f wmma_bf16(v16bf a, v16bf b, v8f c) {
    return __builtin_amdgcn_wmma_f32_16x16x32_bf16(
        /*neg_a=*/false, a, /*neg_b=*/false, b,
        /*c_mod=*/(short)0, c, /*reuse_a=*/false, /*reuse_b=*/false);
}

// Per-lane 16B global->LDS copy: async (GLOBAL_LOAD_ASYNC_TO_LDS_B128) when
// available, otherwise via VGPRs.
static __device__ inline void copy16_g2l(const bf16_t* g, bf16_t* l) {
#if HAVE_ASYNC_LDS
    __builtin_amdgcn_global_load_async_to_lds_b128(
        (__attribute__((address_space(1))) v4i_t*)g,
        (__attribute__((address_space(3))) v4i_t*)l, 0, 0);
#else
    *(v8bf*)l = *(const v8bf*)g;
#endif
}
static __device__ inline void async_wait() {
#if HAVE_ASYNC_LDS
    __builtin_amdgcn_s_wait_asynccnt(0);
#endif
}

// Load a 16-lane-row x 32-k bf16 fragment (A or symmetric-B layout):
// lane holds row/col = lane&15; element e maps to k = (e&7)+(e>=8?16:0)+hi*8.
static __device__ inline v16bf load_frag(const bf16_t* base, int hi) {
    v8bf lo = *(const v8bf*)(base + hi * 8);
    v8bf up = *(const v8bf*)(base + 16 + hi * 8);
    v16bf r;
#pragma unroll
    for (int i = 0; i < 8; ++i) { r[i] = lo[i]; r[i + 8] = up[i]; }
    return r;
}

// ---------------------------------------------------------------------------
// Problem constants
// ---------------------------------------------------------------------------
#define D_MODEL 1024
#define N_HEADS 16
#define KV_HEADS 4
#define D_HEAD 64
#define SEQ 2048
#define BATCH 2
#define M_TOT (BATCH * SEQ)   // 4096

// ---------------------------------------------------------------------------
// Kernel 1: RMSNorm (f32 -> bf16), one block per row of 1024
// ---------------------------------------------------------------------------
__global__ void rmsnorm_bf16_kernel(const float* __restrict__ x,
                                    const float* __restrict__ g,
                                    bf16_t* __restrict__ xn) {
    const int row = blockIdx.x;
    const float* xr = x + (size_t)row * D_MODEL;
    int tid = threadIdx.x, lane = tid & 31, w = tid >> 5;

    float ss = 0.f;
    for (int i = tid; i < D_MODEL; i += 256) { float v = xr[i]; ss += v * v; }
#pragma unroll
    for (int off = 16; off >= 1; off >>= 1) ss += __shfl_xor(ss, off, 32);

    __shared__ float red[8];
    if (lane == 0) red[w] = ss;
    __syncthreads();
    if (w == 0) {
        float t = (lane < 8) ? red[lane] : 0.f;
#pragma unroll
        for (int off = 4; off >= 1; off >>= 1) t += __shfl_xor(t, off, 8);
        if (lane == 0) red[0] = t;
    }
    __syncthreads();
    const float inv = rsqrtf(red[0] * (1.0f / (float)D_MODEL) + 1.1920928955078125e-07f);
    for (int i = tid; i < D_MODEL; i += 256)
        xn[(size_t)row * D_MODEL + i] = f2bf(xr[i] * inv * g[i]);
}

// ---------------------------------------------------------------------------
// Kernel 2: f32 -> bf16 conversion
// ---------------------------------------------------------------------------
__global__ void cvt_bf16_kernel(const float* __restrict__ in, bf16_t* __restrict__ out, int n) {
    int i = blockIdx.x * 256 + threadIdx.x;
    if (i < n) out[i] = f2bf(in[i]);
}

// ---------------------------------------------------------------------------
// Kernel 3: tiled bf16 GEMM, C = A[M,K] @ B[K,N].  Tile 128x128x32, 8 waves,
// wave tile 32x64 (2x4 WMMA).  Double-buffered LDS, software-pipelined:
// tile k+1 is fetched (async-to-LDS for A) while WMMA consumes tile k.
// ---------------------------------------------------------------------------
#define BM 128
#define BN 128
#define BK 32
#define LDA (BK + 8)   // 40 elems = 80B rows, 16B-aligned

__global__ void gemm_bf16_kernel(const bf16_t* __restrict__ A,
                                 const bf16_t* __restrict__ B,
                                 bf16_t* __restrict__ Cb,
                                 float* __restrict__ Cf,
                                 const float* __restrict__ res,
                                 int M, int N, int K) {
    __shared__ __align__(16) bf16_t As[2][BM][LDA];   // row-major A tiles
    __shared__ __align__(16) bf16_t Bst[2][BN][LDA];  // B tiles transposed [n][k]

    const int tid = threadIdx.x;
    const int lane = tid & 31, w = tid >> 5;
    const int wm = w >> 1, wn = w & 1;       // 4 (M) x 2 (N) wave grid
    const int lm = lane & 15, hi = lane >> 4;
    const int m0 = blockIdx.y * BM;
    const int n0 = blockIdx.x * BN;

    // Per-thread staging coordinates (fixed all kernel)
    const int ar_ = tid >> 2, ac_ = (tid & 3) * 8;    // A: 64 rows/pass, 4x8 cols
    const int bk_ = tid >> 4, bc_ = (tid & 15) * 8;   // B: 16 k/pass, 16x8 cols
    const bf16_t* aSrc0 = A + (size_t)(m0 + ar_) * K + ac_;
    const bf16_t* aSrc1 = aSrc0 + (size_t)64 * K;
    const bf16_t* bSrc0 = B + (size_t)bk_ * N + n0 + bc_;
    const bf16_t* bSrc1 = bSrc0 + (size_t)16 * N;

    v8f acc[2][4];
#pragma unroll
    for (int i = 0; i < 2; ++i)
#pragma unroll
        for (int j = 0; j < 4; ++j)
#pragma unroll
            for (int r = 0; r < 8; ++r) acc[i][j][r] = 0.f;

    // ---- prologue: stage tile 0 into buffer 0
    {
#if HAVE_ASYNC_LDS
        copy16_g2l(aSrc0, &As[0][ar_][ac_]);
        copy16_g2l(aSrc1, &As[0][ar_ + 64][ac_]);
#else
        v8bf a0 = *(const v8bf*)aSrc0;
        v8bf a1 = *(const v8bf*)aSrc1;
        *(v8bf*)&As[0][ar_][ac_] = a0;
        *(v8bf*)&As[0][ar_ + 64][ac_] = a1;
#endif
        v8bf b0 = *(const v8bf*)bSrc0;
        v8bf b1 = *(const v8bf*)bSrc1;
#pragma unroll
        for (int j = 0; j < 8; ++j) {
            Bst[0][bc_ + j][bk_]      = b0[j];
            Bst[0][bc_ + j][bk_ + 16] = b1[j];
        }
        async_wait();
    }
    __syncthreads();
    aSrc0 += BK; aSrc1 += BK;
    bSrc0 += (size_t)BK * N; bSrc1 += (size_t)BK * N;

    int buf = 0;
    for (int kt = 0; kt < K; kt += BK) {
        const bool more = (kt + BK) < K;
        const int nbuf = buf ^ 1;

        // ---- issue prefetch of next tile (overlaps with WMMA below)
        v8bf b0v, b1v;
#if !HAVE_ASYNC_LDS
        v8bf a0v, a1v;
#endif
        if (more) {
#if HAVE_ASYNC_LDS
            copy16_g2l(aSrc0, &As[nbuf][ar_][ac_]);
            copy16_g2l(aSrc1, &As[nbuf][ar_ + 64][ac_]);
#else
            a0v = *(const v8bf*)aSrc0;
            a1v = *(const v8bf*)aSrc1;
#endif
            b0v = *(const v8bf*)bSrc0;
            b1v = *(const v8bf*)bSrc1;
            if (kt + 2 * BK < K) {   // look 2 tiles ahead into L2/WGP$
                __builtin_prefetch(aSrc0 + BK, 0, 1);
                __builtin_prefetch(bSrc0 + (size_t)BK * N, 0, 1);
            }
        }

        // ---- WMMA on current buffer
        v16bf af[2], bfr[4];
#pragma unroll
        for (int i = 0; i < 2; ++i)
            af[i] = load_frag(&As[buf][wm * 32 + i * 16 + lm][0], hi);
#pragma unroll
        for (int j = 0; j < 4; ++j)
            bfr[j] = load_frag(&Bst[buf][wn * 64 + j * 16 + lm][0], hi);
#pragma unroll
        for (int i = 0; i < 2; ++i)
#pragma unroll
            for (int j = 0; j < 4; ++j)
                acc[i][j] = wmma_bf16(af[i], bfr[j], acc[i][j]);

        // ---- deposit prefetched data into next buffer
        if (more) {
#if !HAVE_ASYNC_LDS
            *(v8bf*)&As[nbuf][ar_][ac_]      = a0v;
            *(v8bf*)&As[nbuf][ar_ + 64][ac_] = a1v;
#endif
#pragma unroll
            for (int j = 0; j < 8; ++j) {
                Bst[nbuf][bc_ + j][bk_]      = b0v[j];
                Bst[nbuf][bc_ + j][bk_ + 16] = b1v[j];
            }
            aSrc0 += BK; aSrc1 += BK;
            bSrc0 += (size_t)BK * N; bSrc1 += (size_t)BK * N;
        }
        async_wait();
        __syncthreads();
        buf = nbuf;
    }

    // Epilogue: C element (m = r + hi*8, n = lane&15) within each 16x16 tile
#pragma unroll
    for (int i = 0; i < 2; ++i)
#pragma unroll
        for (int j = 0; j < 4; ++j)
#pragma unroll
            for (int r = 0; r < 8; ++r) {
                int rr = m0 + wm * 32 + i * 16 + r + hi * 8;
                int cc = n0 + wn * 64 + j * 16 + lm;
                float v = acc[i][j][r];
                if (res) v += res[(size_t)rr * N + cc];
                if (Cf) Cf[(size_t)rr * N + cc] = v;
                if (Cb) Cb[(size_t)rr * N + cc] = f2bf(v);
            }
}

// ---------------------------------------------------------------------------
// Kernel 4: RoPE factor + repack [4096, H*64] -> [b][H][2048][64]
// ---------------------------------------------------------------------------
__global__ void rope_pack_kernel(const bf16_t* __restrict__ in,
                                 bf16_t* __restrict__ out,
                                 int heads, int applyRf) {
    const int C = heads * D_HEAD;
    size_t i = (size_t)blockIdx.x * 256 + threadIdx.x;
    if (i >= (size_t)M_TOT * C) return;
    int m = (int)(i / C), c = (int)(i % C);
    int b = m >> 11, n = m & (SEQ - 1);
    int hh = c >> 6, d = c & 63;
    float v = bf2f(in[i]);
    if (applyRf) {
        float fe = -2.0f * (float)(d >> 1) * (1.0f / 64.0f);
        float theta = __powf(10000.0f, fe);
        float ang = (float)n * theta;
        float sgn = (n & 1) ? -1.0f : 1.0f;
        v *= (cosf(ang) + sgn * sinf(ang));
    }
    out[(((size_t)b * heads + hh) * SEQ + n) * D_HEAD + d] = f2bf(v);
}

// ---------------------------------------------------------------------------
// Kernel 5: flash attention.  Grid: (b*16 heads, 2048/128 query blocks).
// 8 waves; wave owns 16 query rows.  K/V blocks of 64 double-buffered in LDS
// (async-to-LDS for K); online softmax; P round-trips per-wave LDS to become
// an A fragment for the P@V WMMA.
// ---------------------------------------------------------------------------
#define KBLK 64
#define LDK 72   // padded stride (144B rows, 16B aligned)

__global__ void flash_attn_kernel(const bf16_t* __restrict__ Qg,
                                  const bf16_t* __restrict__ Kg,
                                  const bf16_t* __restrict__ Vg,
                                  float* __restrict__ Sf,
                                  bf16_t* __restrict__ Sb) {
    __shared__ __align__(16) bf16_t Ks[2][KBLK][LDK];    // [key][feat]
    __shared__ __align__(16) bf16_t Vt[2][D_HEAD][LDK];  // [feat][key]
    __shared__ __align__(16) bf16_t Ps[8][16][LDK];      // per-wave P tile

    const int tid = threadIdx.x;
    const int lane = tid & 31, w = tid >> 5;
    const int lm = lane & 15, hi = lane >> 4;
    const int bh = blockIdx.x;
    const int b = bh >> 4, h = bh & 15, kvh = h & (KV_HEADS - 1);
    const int q0 = blockIdx.y * 128;

    const bf16_t* Qbase = Qg + (size_t)(b * N_HEADS + h) * SEQ * D_HEAD;
    const bf16_t* Kbase = Kg + (size_t)(b * KV_HEADS + kvh) * SEQ * D_HEAD;
    const bf16_t* Vbase = Vg + (size_t)(b * KV_HEADS + kvh) * SEQ * D_HEAD;

    // Per-thread staging coordinates: 32 rows/pass, 8x8 cols
    const int kr_ = tid >> 3, kc_ = (tid & 7) * 8;
    const bf16_t* kSrc0 = Kbase + (size_t)kr_ * D_HEAD + kc_;
    const bf16_t* kSrc1 = kSrc0 + 32 * D_HEAD;
    const bf16_t* vSrc0 = Vbase + (size_t)kr_ * D_HEAD + kc_;
    const bf16_t* vSrc1 = vSrc0 + 32 * D_HEAD;

    // Q fragments for this wave's 16 rows (held in registers all kernel)
    v16bf qf[2];
#pragma unroll
    for (int ks = 0; ks < 2; ++ks)
        qf[ks] = load_frag(Qbase + (size_t)(q0 + w * 16 + lm) * D_HEAD + ks * 32, hi);

    float mrow[8], lrow[8];
    v8f oacc[4];
#pragma unroll
    for (int r = 0; r < 8; ++r) { mrow[r] = -1e30f; lrow[r] = 0.f; }
#pragma unroll
    for (int ft = 0; ft < 4; ++ft)
#pragma unroll
        for (int r = 0; r < 8; ++r) oacc[ft][r] = 0.f;

    // ---- prologue: stage block 0 into buffer 0
    {
#if HAVE_ASYNC_LDS
        copy16_g2l(kSrc0, &Ks[0][kr_][kc_]);
        copy16_g2l(kSrc1, &Ks[0][kr_ + 32][kc_]);
#else
        v8bf k0 = *(const v8bf*)kSrc0;
        v8bf k1 = *(const v8bf*)kSrc1;
        *(v8bf*)&Ks[0][kr_][kc_]      = k0;
        *(v8bf*)&Ks[0][kr_ + 32][kc_] = k1;
#endif
        v8bf t0 = *(const v8bf*)vSrc0;
        v8bf t1 = *(const v8bf*)vSrc1;
#pragma unroll
        for (int j = 0; j < 8; ++j) {
            Vt[0][kc_ + j][kr_]      = t0[j];
            Vt[0][kc_ + j][kr_ + 32] = t1[j];
        }
        async_wait();
    }
    __syncthreads();
    kSrc0 += KBLK * D_HEAD; kSrc1 += KBLK * D_HEAD;
    vSrc0 += KBLK * D_HEAD; vSrc1 += KBLK * D_HEAD;

    int buf = 0;
    for (int kb = 0; kb < SEQ; kb += KBLK) {
        const bool more = (kb + KBLK) < SEQ;
        const int nbuf = buf ^ 1;

        // ---- issue prefetch of next K/V block
        v8bf v0v, v1v;
#if !HAVE_ASYNC_LDS
        v8bf k0v, k1v;
#endif
        if (more) {
#if HAVE_ASYNC_LDS
            copy16_g2l(kSrc0, &Ks[nbuf][kr_][kc_]);
            copy16_g2l(kSrc1, &Ks[nbuf][kr_ + 32][kc_]);
#else
            k0v = *(const v8bf*)kSrc0;
            k1v = *(const v8bf*)kSrc1;
#endif
            v0v = *(const v8bf*)vSrc0;
            v1v = *(const v8bf*)vSrc1;
        }

        // ---- S = (Q K^T) / sqrt(d_model), 4 column tiles of 16 keys
        v8f sc[4];
#pragma unroll
        for (int ct = 0; ct < 4; ++ct) {
            v8f s = {0.f, 0.f, 0.f, 0.f, 0.f, 0.f, 0.f, 0.f};
#pragma unroll
            for (int ks = 0; ks < 2; ++ks) {
                v16bf kf = load_frag(&Ks[buf][ct * 16 + lm][ks * 32], hi);
                s = wmma_bf16(qf[ks], kf, s);
            }
#pragma unroll
            for (int r = 0; r < 8; ++r) s[r] *= 0.03125f;   // 1/sqrt(1024)
            sc[ct] = s;
        }

        // ---- online softmax (row m = r + hi*8 lives across a 16-lane half)
        float alpha[8], rs[8];
#pragma unroll
        for (int r = 0; r < 8; ++r) {
            float v = fmaxf(fmaxf(sc[0][r], sc[1][r]), fmaxf(sc[2][r], sc[3][r]));
#pragma unroll
            for (int off = 8; off >= 1; off >>= 1) v = fmaxf(v, __shfl_xor(v, off, 16));
            float mnew = fmaxf(mrow[r], v);
            alpha[r] = __expf(mrow[r] - mnew);
            mrow[r] = mnew;
            rs[r] = 0.f;
        }
#pragma unroll
        for (int ct = 0; ct < 4; ++ct)
#pragma unroll
            for (int r = 0; r < 8; ++r) {
                float pv = __expf(sc[ct][r] - mrow[r]);
                rs[r] += pv;
                Ps[w][r + hi * 8][ct * 16 + lm] = f2bf(pv);   // C-layout -> LDS
            }
#pragma unroll
        for (int r = 0; r < 8; ++r) {
            float t = rs[r];
#pragma unroll
            for (int off = 8; off >= 1; off >>= 1) t += __shfl_xor(t, off, 16);
            lrow[r] = lrow[r] * alpha[r] + t;
        }
#pragma unroll
        for (int ft = 0; ft < 4; ++ft)
#pragma unroll
            for (int r = 0; r < 8; ++r) oacc[ft][r] *= alpha[r];

        // ---- O += P @ V
#pragma unroll
        for (int ks2 = 0; ks2 < 2; ++ks2) {
            v16bf pf = load_frag(&Ps[w][lm][ks2 * 32], hi);
#pragma unroll
            for (int ft = 0; ft < 4; ++ft) {
                v16bf vf = load_frag(&Vt[buf][ft * 16 + lm][ks2 * 32], hi);
                oacc[ft] = wmma_bf16(pf, vf, oacc[ft]);
            }
        }

        // ---- deposit prefetched block into next buffer
        if (more) {
#if !HAVE_ASYNC_LDS
            *(v8bf*)&Ks[nbuf][kr_][kc_]      = k0v;
            *(v8bf*)&Ks[nbuf][kr_ + 32][kc_] = k1v;
#endif
#pragma unroll
            for (int j = 0; j < 8; ++j) {
                Vt[nbuf][kc_ + j][kr_]      = v0v[j];
                Vt[nbuf][kc_ + j][kr_ + 32] = v1v[j];
            }
            kSrc0 += KBLK * D_HEAD; kSrc1 += KBLK * D_HEAD;
            vSrc0 += KBLK * D_HEAD; vSrc1 += KBLK * D_HEAD;
        }
        async_wait();
        __syncthreads();
        buf = nbuf;
    }

    // ---- normalize and write s as [b][n][h][d] (f32 + bf16)
#pragma unroll
    for (int ft = 0; ft < 4; ++ft)
#pragma unroll
        for (int r = 0; r < 8; ++r) {
            int qi = q0 + w * 16 + r + hi * 8;
            int d = ft * 16 + lm;
            size_t o = (((size_t)b * SEQ + qi) * N_HEADS + h) * D_HEAD + d;
            float val = oacc[ft][r] / lrow[r];
            Sf[o] = val;
            Sb[o] = f2bf(val);
        }
}

// ---------------------------------------------------------------------------
// Host launch
// ---------------------------------------------------------------------------
extern "C" void kernel_launch(void* const* d_in, const int* in_sizes, int n_in,
                              void* d_out, int out_size, void* d_ws, size_t ws_size,
                              hipStream_t stream) {
    (void)in_sizes; (void)n_in; (void)out_size; (void)ws_size;
    const float* x   = (const float*)d_in[0];
    const float* w_q = (const float*)d_in[1];
    const float* w_k = (const float*)d_in[2];
    const float* w_v = (const float*)d_in[3];
    const float* w_o = (const float*)d_in[4];
    const float* g   = (const float*)d_in[5];
    float* out = (float*)d_out;

    char* ws = (char*)d_ws;
    bf16_t* xn   = (bf16_t*)(ws + 0);          //  8.0 MB  x_norm bf16 [4096,1024]
    bf16_t* wqb  = (bf16_t*)(ws + 8388608);    //  2.0 MB
    bf16_t* wkb  = (bf16_t*)(ws + 10485760);   //  0.5 MB
    bf16_t* wvb  = (bf16_t*)(ws + 11010048);   //  0.5 MB
    bf16_t* wob  = (bf16_t*)(ws + 11534336);   //  2.0 MB
    bf16_t* qtmp = (bf16_t*)(ws + 13631488);   //  8.0 MB  q gemm out [4096,1024]
    bf16_t* ktmp = (bf16_t*)(ws + 22020096);   //  2.0 MB  [4096,256]
    bf16_t* vtmp = (bf16_t*)(ws + 24117248);   //  2.0 MB
    bf16_t* Qp   = (bf16_t*)(ws + 26214400);   //  8.0 MB  [b][16][2048][64]
    bf16_t* Kp   = (bf16_t*)(ws + 34603008);   //  2.0 MB  [b][4][2048][64]
    bf16_t* Vp   = (bf16_t*)(ws + 36700160);   //  2.0 MB
    float*  sF   = (float*) (ws + 38797312);   // 16.0 MB  s f32 [b,n,1024]
    bf16_t* sB   = (bf16_t*)(ws + 55574528);   //  8.0 MB  s bf16

    // 1. RMSNorm
    rmsnorm_bf16_kernel<<<M_TOT, 256, 0, stream>>>(x, g, xn);

    // 2. Weight conversion
    cvt_bf16_kernel<<<4096, 256, 0, stream>>>(w_q, wqb, D_MODEL * D_MODEL);
    cvt_bf16_kernel<<<1024, 256, 0, stream>>>(w_k, wkb, D_MODEL * D_HEAD * KV_HEADS);
    cvt_bf16_kernel<<<1024, 256, 0, stream>>>(w_v, wvb, D_MODEL * D_HEAD * KV_HEADS);
    cvt_bf16_kernel<<<4096, 256, 0, stream>>>(w_o, wob, D_MODEL * D_MODEL);

    // 3. Q/K/V projections (bf16 WMMA GEMM)
    gemm_bf16_kernel<<<dim3(D_MODEL / BN, M_TOT / BM), 256, 0, stream>>>(
        xn, wqb, qtmp, nullptr, nullptr, M_TOT, D_MODEL, D_MODEL);
    gemm_bf16_kernel<<<dim3(256 / BN, M_TOT / BM), 256, 0, stream>>>(
        xn, wkb, ktmp, nullptr, nullptr, M_TOT, 256, D_MODEL);
    gemm_bf16_kernel<<<dim3(256 / BN, M_TOT / BM), 256, 0, stream>>>(
        xn, wvb, vtmp, nullptr, nullptr, M_TOT, 256, D_MODEL);

    // 4. RoPE + head-major repack
    rope_pack_kernel<<<(M_TOT * D_MODEL) / 256, 256, 0, stream>>>(qtmp, Qp, N_HEADS, 1);
    rope_pack_kernel<<<(M_TOT * 256) / 256, 256, 0, stream>>>(ktmp, Kp, KV_HEADS, 1);
    rope_pack_kernel<<<(M_TOT * 256) / 256, 256, 0, stream>>>(vtmp, Vp, KV_HEADS, 0);

    // 5. Flash attention
    flash_attn_kernel<<<dim3(BATCH * N_HEADS, SEQ / 128), 256, 0, stream>>>(
        Qp, Kp, Vp, sF, sB);

    // 6. out = s + s @ w_o  (WMMA GEMM with f32 residual epilogue)
    gemm_bf16_kernel<<<dim3(D_MODEL / BN, M_TOT / BM), 256, 0, stream>>>(
        sB, wob, nullptr, out, sF, M_TOT, D_MODEL, D_MODEL);
}
